// DEQTransformerLM_74217034875081
// MI455X (gfx1250) — matrix operand order repaired and checked by reading, not compiled, not run on hardware
//
#include <hip/hip_runtime.h>

typedef __attribute__((ext_vector_type(16))) __bf16 v16bf;
typedef __attribute__((ext_vector_type(8)))  float  v8f;

constexpr int CB  = 2;      // batch
constexpr int CD  = 1024;   // d_model
constexpr int CQ  = 1024;   // qlen
constexpr int CM  = 1024;   // mlen
constexpr int CK  = 2048;   // klen = qlen + mlen
constexpr int CNH = 16;     // heads
constexpr int CDH = 64;     // head dim
constexpr int CDI = 4096;   // ffn inner
constexpr int CLOCAL = 1000;

// ---------- bf16 helpers (bit-level, RNE) ----------
__device__ __forceinline__ unsigned short f2bf(float f) {
  unsigned u = __float_as_uint(f);
  u += 0x7FFFu + ((u >> 16) & 1u);
  return (unsigned short)(u >> 16);
}
__device__ __forceinline__ float bf2f(unsigned short h) {
  return __uint_as_float(((unsigned)h) << 16);
}

union FragBF { v16bf v; uint4 q[2]; };

// A fragment: 16x32 bf16 (ISA 7.12.2): lane row=lane%16, half h=lane/16,
// elems 0..7 -> k = 8h+e ; elems 8..15 -> k = 16+8h+(e-8)
__device__ __forceinline__ v16bf load_a_frag(const unsigned short* a, int lda,
                                             int row0, int k0, int lane) {
  const int h = lane >> 4;
  const unsigned short* p = a + (long long)(row0 + (lane & 15)) * lda + k0;
  FragBF f;
  f.q[0] = *(const uint4*)(p + 8 * h);
  f.q[1] = *(const uint4*)(p + 16 + 8 * h);
  return f.v;
}
// B fragment: 32x16 bf16: lane col=lane%16, half h=lane/16,
// elems 0..7 -> k = 8h+e ; elems 8..15 -> k = 16+8h+(e-8)
__device__ __forceinline__ v16bf load_b_frag(const unsigned short* b, int ldb,
                                             int col0, int k0, int lane) {
  const unsigned short* p =
      b + (long long)(col0 + (lane & 15)) * ldb + k0 + 8 * (lane >> 4);
  FragBF f;
  f.q[0] = *(const uint4*)(p);
  f.q[1] = *(const uint4*)(p + 16);
  return f.v;
}

__device__ __forceinline__ v8f wmma_bf16(v16bf a, v16bf b, v8f c) {
  return __builtin_amdgcn_wmma_f32_16x16x32_bf16(false, a, false, b,
                                                 (short)0, c, false, false);
}

// ---------- elementwise f32 -> bf16 ----------
__global__ __launch_bounds__(256) void k_cvt(const float* in, unsigned short* out, int n) {
  int i = blockIdx.x * 256 + threadIdx.x;
  if (i < n) out[i] = f2bf(in[i]);
}

// ---------- tiled transpose (f32|bf16 in) -> bf16 out ----------
__global__ __launch_bounds__(256) void k_transpose(const void* in, int in_is_f32,
                                                   int R, int C, int ldin,
                                                   unsigned short* out, int ldout) {
  __shared__ float t[32][33];
  const int c0 = blockIdx.x * 32, r0 = blockIdx.y * 32;
  const int tx = threadIdx.x, ty = threadIdx.y;  // block (32, 8)
#pragma unroll
  for (int i = 0; i < 4; ++i) {
    int r = r0 + ty + i * 8, c = c0 + tx;
    float v = 0.f;
    if (r < R && c < C)
      v = in_is_f32 ? ((const float*)in)[(long long)r * ldin + c]
                    : bf2f(((const unsigned short*)in)[(long long)r * ldin + c]);
    t[ty + i * 8][tx] = v;
  }
  __syncthreads();
#pragma unroll
  for (int i = 0; i < 4; ++i) {
    int orow = c0 + ty + i * 8, oc = r0 + tx;
    if (orow < C && oc < R)
      out[(long long)orow * ldout + oc] = f2bf(t[tx][ty + i * 8]);
  }
}

// ---------- generic batched WMMA GEMM ----------
// C[m,n] = sum_k A[m,k] * B[n,k]  (A,B bf16 row-major, K-contiguous)
// batches: z -> bb = z/nh, nn = z%nh ; offsets = bb*SB + nn*SN (elements)
// epilogues: addNM[n*ld+m], addMN[m*ld+n], biasN[n], relu; out bf16 or f32.
__global__ __launch_bounds__(256) void k_gemm(
    const unsigned short* A, const unsigned short* Bm,
    long long aSB, long long aSN, int lda,
    long long bSB, long long bSN, int ldb,
    int Msz, int Nsz, int Ksz, int nh,
    const float* addNM, int addNMld, long long addNMSB,
    const float* addMN, int addMNld, long long addMNSB,
    const float* biasN, int relu,
    unsigned short* outBF, float* outF, int ldc, long long cSB, long long cSN) {
  (void)Msz;
  const int z = blockIdx.z;
  const int bb = z / nh, nn = z % nh;
  const unsigned short* Ab = A + bb * aSB + nn * aSN;
  const unsigned short* Bb = Bm + bb * bSB + nn * bSN;
  const long long cOff = bb * cSB + nn * cSN;

  // wave index forced into an SGPR: all tile guards below become scalar
  // branches, so EXEC stays all-1s across every WMMA (ISA 7.12 requirement).
  const int w = __builtin_amdgcn_readfirstlane((int)(threadIdx.x >> 5));
  const int lane = threadIdx.x & 31;
  const int wm = w & 3, wn = w >> 2;
  const int rowBase = blockIdx.x * 128 + wm * 32;
  const int colBase = blockIdx.y * 128 + wn * 64;
  // number of active 16-wide column tiles for this wave (scalar, 0..4)
  int ntAct = 0;
#pragma unroll
  for (int nt = 0; nt < 4; ++nt)
    if (colBase + nt * 16 < Nsz) ntAct = nt + 1;

  const v8f vzero = {0.f, 0.f, 0.f, 0.f, 0.f, 0.f, 0.f, 0.f};
  v8f acc[2][4];
#pragma unroll
  for (int i = 0; i < 2; ++i)
#pragma unroll
    for (int j = 0; j < 4; ++j) acc[i][j] = vzero;

  for (int k0 = 0; k0 < Ksz; k0 += 32) {
    // Load A fragments and ALL active B fragments first (one load clause),
    // then issue the 8 WMMAs back-to-back: one wait per k-step instead of
    // one wait per tile, and hazard NOPs amortize over the whole block.
    v16bf a0 = load_a_frag(Ab, lda, rowBase, k0, lane);
    v16bf a1 = load_a_frag(Ab, lda, rowBase + 16, k0, lane);
    v16bf bf[4];
#pragma unroll
    for (int nt = 0; nt < 4; ++nt)
      if (nt < ntAct) bf[nt] = load_b_frag(Bb, ldb, colBase + nt * 16, k0, lane);
#pragma unroll
    for (int nt = 0; nt < 4; ++nt)
      if (nt < ntAct) {
        acc[0][nt] = wmma_bf16(a0, bf[nt], acc[0][nt]);
        acc[1][nt] = wmma_bf16(a1, bf[nt], acc[1][nt]);
      }
  }

  const int h = lane >> 4, cl = lane & 15;
#pragma unroll
  for (int mt = 0; mt < 2; ++mt)
#pragma unroll
    for (int nt = 0; nt < 4; ++nt) {
      if (nt >= ntAct) continue;
      int col = colBase + nt * 16 + cl;
#pragma unroll
      for (int r = 0; r < 8; ++r) {
        int row = rowBase + mt * 16 + r + 8 * h;
        float v = acc[mt][nt][r];
        if (addNM) v += addNM[bb * addNMSB + (long long)col * addNMld + row];
        if (addMN) v += addMN[bb * addMNSB + (long long)row * addMNld + col];
        if (biasN) v += biasN[col];
        if (relu) v = fmaxf(v, 0.f);
        long long o = cOff + (long long)row * ldc + col;
        if (outBF) outBF[o] = f2bf(v);
        else       outF[o] = v;
      }
    }
}

// ---------- q + r_w_bias / q + r_r_bias (bf16) ----------
__global__ __launch_bounds__(256) void k_qbias(const unsigned short* qkv,
                                               const float* rwb, const float* rrb,
                                               unsigned short* qw, unsigned short* qr) {
  long long idx = (long long)blockIdx.x * 256 + threadIdx.x;  // over B*Q*D
  int c = (int)(idx % CD);
  long long t = idx / CD;
  int i = (int)(t % CQ);
  int b = (int)(t / CQ);
  float v = bf2f(qkv[((long long)b * CK + CM + i) * (3 * CD) + c]);
  qw[idx] = f2bf(v + rwb[c]);
  qr[idx] = f2bf(v + rrb[c]);
}

// ---------- fused AC-GEMM + rel-shifted BD + mask + softmax ----------
// block: 16 query rows x full 2048 keys for one (b, head). 8 waves.
__global__ __launch_bounds__(256) void k_score_softmax(const unsigned short* qw,
                                                       const unsigned short* qkv,
                                                       const unsigned short* BD,
                                                       unsigned short* P) {
  extern __shared__ float s[];          // [16][CK] scores
  __shared__ float red[256];
  const int qb = blockIdx.x, n = blockIdx.y, b = blockIdx.z;
  const int tid = threadIdx.x;
  const int w = __builtin_amdgcn_readfirstlane((int)(threadIdx.x >> 5));
  const int lane = tid & 31;
  const int rowBase = qb * 16;

  const unsigned short* Aq = qw + (long long)b * CQ * CD + (long long)n * CDH;
  const unsigned short* Bk = qkv + (long long)b * CK * (3 * CD) + CD + (long long)n * CDH;
  const unsigned short* bd = BD + ((long long)(b * CNH + n)) * CQ * CK;

  v16bf a0 = load_a_frag(Aq, CD, rowBase, 0, lane);
  v16bf a1 = load_a_frag(Aq, CD, rowBase, 32, lane);
  const v8f vzero = {0.f, 0.f, 0.f, 0.f, 0.f, 0.f, 0.f, 0.f};
  const int h = lane >> 4, cl = lane & 15;

  // two column tiles per iteration: 8 b128 loads in flight per wait
  for (int t = 0; t < 16; t += 2) {
    int colA = (w + 8 * t) * 16;          // scalar (w is SGPR)
    int colB = (w + 8 * (t + 1)) * 16;
    v16bf b0 = load_b_frag(Bk, 3 * CD, colA, 0, lane);
    v16bf b1 = load_b_frag(Bk, 3 * CD, colA, 32, lane);
    v16bf b2 = load_b_frag(Bk, 3 * CD, colB, 0, lane);
    v16bf b3 = load_b_frag(Bk, 3 * CD, colB, 32, lane);
    v8f accA = vzero, accB = vzero;
    accA = wmma_bf16(a0, b0, accA);
    accB = wmma_bf16(a0, b2, accB);
    accA = wmma_bf16(a1, b1, accA);
    accB = wmma_bf16(a1, b3, accB);
#pragma unroll
    for (int half = 0; half < 2; ++half) {
      int col0 = half ? colB : colA;
      const v8f& acc = half ? accB : accA;
      int col = col0 + cl;
#pragma unroll
      for (int r = 0; r < 8; ++r) {
        int ri = r + 8 * h;
        int i = rowBase + ri;                     // global query index
        // Transformer-XL rel-shift: out[i,j] = padded_flat[i*K + j + Q]
        long long f = (long long)i * CK + col + CQ;
        int ry = (int)(f / (CK + 1));
        int cy = (int)(f % (CK + 1));
        float bdv = cy ? bf2f(bd[(long long)ry * CK + (cy - 1)]) : 0.f;
        bool masked = (col > i + CM) || (col <= i + (CM - CLOCAL));
        float v = masked ? -__builtin_inff() : (acc[r] + bdv) * 0.125f; // 1/sqrt(64)
        s[ri * CK + col] = v;
      }
    }
  }
  __syncthreads();

  unsigned short* Pout =
      P + ((long long)(b * CNH + n)) * CQ * CK + (long long)rowBase * CK;
  for (int r = 0; r < 16; ++r) {
    float m = -__builtin_inff();
    for (int j = tid; j < CK; j += 256) m = fmaxf(m, s[r * CK + j]);
    red[tid] = m; __syncthreads();
    for (int st = 128; st > 0; st >>= 1) {
      if (tid < st) red[tid] = fmaxf(red[tid], red[tid + st]);
      __syncthreads();
    }
    m = red[0]; __syncthreads();
    float sum = 0.f;
    for (int j = tid; j < CK; j += 256) {
      float e = __expf(s[r * CK + j] - m);
      s[r * CK + j] = e;
      sum += e;
    }
    red[tid] = sum; __syncthreads();
    for (int st = 128; st > 0; st >>= 1) {
      if (tid < st) red[tid] += red[tid + st];
      __syncthreads();
    }
    float inv = 1.f / red[0]; __syncthreads();
    for (int j = tid; j < CK; j += 256)
      Pout[(long long)r * CK + j] = f2bf(s[r * CK + j] * inv);
  }
}

// ---------- layernorm over last dim; f32 + bf16 outputs ----------
__global__ __launch_bounds__(256) void k_layernorm(const float* in, float* outF,
                                                   unsigned short* outB, int C) {
  __shared__ float red[256];
  const int row = blockIdx.x, tid = threadIdx.x;
  const float* x = in + (long long)row * C;
  float sm = 0.f;
  for (int j = tid; j < C; j += 256) sm += x[j];
  red[tid] = sm; __syncthreads();
  for (int st = 128; st > 0; st >>= 1) { if (tid < st) red[tid] += red[tid + st]; __syncthreads(); }
  float mean = red[0] / C; __syncthreads();
  float vs = 0.f;
  for (int j = tid; j < C; j += 256) { float d = x[j] - mean; vs += d * d; }
  red[tid] = vs; __syncthreads();
  for (int st = 128; st > 0; st >>= 1) { if (tid < st) red[tid] += red[tid + st]; __syncthreads(); }
  float inv = rsqrtf(red[0] / C + 1e-5f); __syncthreads();
  for (int j = tid; j < C; j += 256) {
    float v = (x[j] - mean) * inv;
    outF[(long long)row * C + j] = v;
    outB[(long long)row * C + j] = f2bf(v);
  }
}

// ---------- final layernorm, writes transposed (B,D,Q) ----------
__global__ __launch_bounds__(256) void k_layernorm_tr(const float* in, float* out,
                                                      int C, int Qd) {
  __shared__ float red[256];
  const int row = blockIdx.x, tid = threadIdx.x;
  const int b = row / Qd, i = row % Qd;
  const float* x = in + (long long)row * C;
  float sm = 0.f;
  for (int j = tid; j < C; j += 256) sm += x[j];
  red[tid] = sm; __syncthreads();
  for (int st = 128; st > 0; st >>= 1) { if (tid < st) red[tid] += red[tid + st]; __syncthreads(); }
  float mean = red[0] / C; __syncthreads();
  float vs = 0.f;
  for (int j = tid; j < C; j += 256) { float d = x[j] - mean; vs += d * d; }
  red[tid] = vs; __syncthreads();
  for (int st = 128; st > 0; st >>= 1) { if (tid < st) red[tid] += red[tid + st]; __syncthreads(); }
  float inv = rsqrtf(red[0] / C + 1e-5f); __syncthreads();
  for (int j = tid; j < C; j += 256) {
    float v = (x[j] - mean) * inv;
    out[((long long)b * C + j) * Qd + i] = v;
  }
}

extern "C" void kernel_launch(void* const* d_in, const int* in_sizes, int n_in,
                              void* d_out, int out_size, void* d_ws, size_t ws_size,
                              hipStream_t stream) {
  (void)in_sizes; (void)n_in; (void)out_size; (void)ws_size;
  typedef long long ll;
  typedef unsigned short bh;

  const float* z      = (const float*)d_in[0];   // (B, D, Q)
  const float* z_hist = (const float*)d_in[1];   // (B, D, M)
  const float* u      = (const float*)d_in[2];   // (B, 3D, K)
  const float* pos    = (const float*)d_in[3];   // (1, D, K)
  const float* Wqkv   = (const float*)d_in[4];   // (3D, D)
  const float* Wr     = (const float*)d_in[5];   // (D, D)
  const float* rwb    = (const float*)d_in[6];   // (NH, DH) flat
  const float* rrb    = (const float*)d_in[7];
  const float* Wo     = (const float*)d_in[8];   // (D, D)
  const float* bo     = (const float*)d_in[9];   // (D)
  const float* Wff1   = (const float*)d_in[10];  // (DI, D)
  const float* bff1   = (const float*)d_in[11];
  const float* Wff2   = (const float*)d_in[12];  // (D, DI)
  const float* bff2   = (const float*)d_in[13];
  float* out = (float*)d_out;                    // (B, D, Q)

  // ---- workspace carve ----
  char* base = (char*)d_ws;
  size_t off = 0;
  auto carve = [&](size_t bytes) -> void* {
    void* p = base + off;
    off = (off + bytes + 255) & ~(size_t)255;
    return p;
  };
  bh* WqkvB = (bh*)carve((size_t)3 * CD * CD * 2);
  bh* WrB   = (bh*)carve((size_t)CD * CD * 2);
  bh* WoB   = (bh*)carve((size_t)CD * CD * 2);
  bh* Wff1B = (bh*)carve((size_t)CDI * CD * 2);
  bh* Wff2B = (bh*)carve((size_t)CD * CDI * 2);
  bh* catT  = (bh*)carve((size_t)CB * CK * CD * 2);   // (B, K, D)
  bh* posT  = (bh*)carve((size_t)CK * CD * 2);        // (K, D)
  bh* qkvB  = (bh*)carve((size_t)CB * CK * 3 * CD * 2); // (B, K, 3D) = w_heads + u
  bh* rkT   = (bh*)carve((size_t)CK * CD * 2);        // (K, D)
  bh* qwB   = (bh*)carve((size_t)CB * CQ * CD * 2);
  bh* qrB   = (bh*)carve((size_t)CB * CQ * CD * 2);
  bh* vT    = (bh*)carve((size_t)CB * CD * CK * 2);   // (B, D, K) v transposed
  bh* BDb   = (bh*)carve((size_t)CB * CNH * CQ * CK * 2);
  bh* Pb    = (bh*)carve((size_t)CB * CNH * CQ * CK * 2);
  bh* avB   = (bh*)carve((size_t)CB * CQ * CD * 2);   // attn_vec token-major
  float* attnOut = (float*)carve((size_t)CB * CQ * CD * 4);
  float* hF      = (float*)carve((size_t)CB * CQ * CD * 4);
  bh*    hB      = (bh*)carve((size_t)CB * CQ * CD * 2);
  bh*    ff1B    = (bh*)carve((size_t)CB * CQ * CDI * 2);
  float* ff2F    = (float*)carve((size_t)CB * CQ * CD * 4);

  const dim3 tb(32, 8);

  // ---- weight converts ----
  k_cvt<<<(3 * CD * CD) / 256, 256, 0, stream>>>(Wqkv, WqkvB, 3 * CD * CD);
  k_cvt<<<(CD * CD) / 256, 256, 0, stream>>>(Wr, WrB, CD * CD);
  k_cvt<<<(CD * CD) / 256, 256, 0, stream>>>(Wo, WoB, CD * CD);
  k_cvt<<<(CDI * CD) / 256, 256, 0, stream>>>(Wff1, Wff1B, CDI * CD);
  k_cvt<<<(CD * CDI) / 256, 256, 0, stream>>>(Wff2, Wff2B, CD * CDI);

  // ---- cat = [z_hist; z] transposed to (B, K, D) bf16 ; pos -> (K, D) ----
  for (int b = 0; b < CB; ++b) {
    k_transpose<<<dim3(CM / 32, CD / 32), tb, 0, stream>>>(
        z_hist + (ll)b * CD * CM, 1, CD, CM, CM, catT + (ll)b * CK * CD, CD);
    k_transpose<<<dim3(CQ / 32, CD / 32), tb, 0, stream>>>(
        z + (ll)b * CD * CQ, 1, CD, CQ, CQ, catT + (ll)b * CK * CD + (ll)CM * CD, CD);
  }
  k_transpose<<<dim3(CK / 32, CD / 32), tb, 0, stream>>>(pos, 1, CD, CK, CK, posT, CD);

  // ---- QKV projection: (B: 2048 x 3072 x 1024), epilogue += u[b,o,k] ----
  k_gemm<<<dim3(CK / 128, (3 * CD) / 128, CB), 256, 0, stream>>>(
      catT, WqkvB,
      (ll)CK * CD, 0, CD,
      0, 0, CD,
      CK, 3 * CD, CD, 1,
      u, CK, (ll)3 * CD * CK,
      nullptr, 0, 0,
      nullptr, 0,
      qkvB, nullptr, 3 * CD, (ll)CK * 3 * CD, 0);

  // ---- r_k = pos @ W_r^T : (2048 x 1024 x 1024) ----
  k_gemm<<<dim3(CK / 128, CD / 128, 1), 256, 0, stream>>>(
      posT, WrB, 0, 0, CD, 0, 0, CD,
      CK, CD, CD, 1,
      nullptr, 0, 0, nullptr, 0, 0, nullptr, 0,
      rkT, nullptr, CD, 0, 0);

  // ---- q + biases ----
  k_qbias<<<(CB * CQ * CD) / 256, 256, 0, stream>>>(qkvB, rwb, rrb, qwB, qrB);

  // ---- v transposed per batch: (K,1024)@ld3072 -> (1024,K) ----
  for (int b = 0; b < CB; ++b)
    k_transpose<<<dim3(CD / 32, CK / 32), tb, 0, stream>>>(
        qkvB + (ll)b * CK * 3 * CD + 2 * CD, 0, CK, CD, 3 * CD,
        vT + (ll)b * CD * CK, CK);

  // ---- BD = (q + r_r_bias) @ r_k^T per (b, head): 1024 x 2048 x 64 ----
  k_gemm<<<dim3(CQ / 128, CK / 128, CB * CNH), 256, 0, stream>>>(
      qrB, rkT,
      (ll)CQ * CD, CDH, CD,
      0, CDH, CD,
      CQ, CK, CDH, CNH,
      nullptr, 0, 0, nullptr, 0, 0, nullptr, 0,
      BDb, nullptr, CK, (ll)CNH * CQ * CK, (ll)CQ * CK);

  // ---- AC + shifted BD + mask + softmax -> prob (bf16) ----
  k_score_softmax<<<dim3(CQ / 16, CNH, CB), 256, (size_t)16 * CK * 4, stream>>>(
      qwB, qkvB, BDb, Pb);

  // ---- attn_vec = prob @ v : per (b, head) 1024 x 64 x 2048 ----
  k_gemm<<<dim3(CQ / 128, 1, CB * CNH), 256, 0, stream>>>(
      Pb, vT,
      (ll)CNH * CQ * CK, (ll)CQ * CK, CK,
      (ll)CD * CK, (ll)CDH * CK, CK,
      CQ, CDH, CK, CNH,
      nullptr, 0, 0, nullptr, 0, 0, nullptr, 0,
      avB, nullptr, CD, (ll)CQ * CD, CDH);

  // ---- attn_out = av @ W_o^T + b_o + z (residual in epilogue), f32 ----
  k_gemm<<<dim3(CQ / 128, CD / 128, CB), 256, 0, stream>>>(
      avB, WoB,
      (ll)CQ * CD, 0, CD,
      0, 0, CD,
      CQ, CD, CD, 1,
      z, CQ, (ll)CD * CQ,
      nullptr, 0, 0,
      bo, 0,
      nullptr, attnOut, CD, (ll)CQ * CD, 0);

  // ---- post-attention layernorm -> h (f32 + bf16) ----
  k_layernorm<<<CB * CQ, 256, 0, stream>>>(attnOut, hF, hB, CD);

  // ---- FFN-1: relu(h @ W_ff1^T + b_ff1), 2048 x 4096 x 1024 ----
  k_gemm<<<dim3((CB * CQ) / 128, CDI / 128, 1), 256, 0, stream>>>(
      hB, Wff1B, 0, 0, CD, 0, 0, CD,
      CB * CQ, CDI, CD, 1,
      nullptr, 0, 0, nullptr, 0, 0,
      bff1, 1,
      ff1B, nullptr, CDI, 0, 0);

  // ---- FFN-2: ff1 @ W_ff2^T + b_ff2 + h (residual), 2048 x 1024 x 4096 ----
  k_gemm<<<dim3((CB * CQ) / 128, CD / 128, 1), 256, 0, stream>>>(
      ff1B, Wff2B, 0, 0, CDI, 0, 0, CDI,
      CB * CQ, CD, CDI, 1,
      nullptr, 0, 0,
      hF, CD, 0,
      bff2, 0,
      nullptr, ff2F, CD, 0, 0);

  // ---- final layernorm, write transposed (B, D, Q) ----
  k_layernorm_tr<<<CB * CQ, 256, 0, stream>>>(ff2F, out, CD, CQ);
}
